// VideoSlotEncoder_84585085927517
// MI455X (gfx1250) — matrix operand order repaired and checked by reading, not compiled, not run on hardware
//
#include <hip/hip_runtime.h>
#include <hip/hip_bf16.h>

// ---------------------------------------------------------------------------
// VideoSlotEncoder for MI455X (gfx1250, wave32, WMMA).
// All big GEMMs: bf16 WMMA (v_wmma_f32_16x16x32_bf16), fp32 accumulate.
// GEMM kernel: 64x128 block tile, 8 waves (4Mx2N), wave tile 16x64
// (4 WMMA per 32-K step), double-buffered LDS, async global->LDS for the
// bf16 weight tile via __builtin_amdgcn_global_load_async_to_lds_b128.
// ---------------------------------------------------------------------------

typedef __attribute__((ext_vector_type(16))) __bf16 v16bf;
typedef __attribute__((ext_vector_type(8)))  float  v8f;

union BfFrag { uint4 u[2]; v16bf f; };

#if __has_builtin(__builtin_amdgcn_global_load_async_to_lds_b128)
#define HAS_ASYNC_LDS 1
#define AS3 __attribute__((address_space(3)))
// first parameter type per clang diagnostic: generic pointer to int __vector(4)
typedef __attribute__((__vector_size__(4 * sizeof(int)))) int async_i32x4;
#endif

__device__ __forceinline__ void async_wait0() {
#if defined(HAS_ASYNC_LDS)
#if __has_builtin(__builtin_amdgcn_s_wait_asynccnt)
  __builtin_amdgcn_s_wait_asynccnt(0);
#else
  asm volatile("s_wait_asynccnt 0" ::: "memory");
#endif
#endif
}

__device__ __forceinline__ unsigned short f2bf_u(float f) {
  unsigned u = __builtin_bit_cast(unsigned, f);
  // round-to-nearest-even to bf16
  return (unsigned short)((u + 0x7FFFu + ((u >> 16) & 1u)) >> 16);
}

// --------------------------- weight conversion -----------------------------
__global__ __launch_bounds__(256) void cvt_f32_bf16(const float* __restrict__ src,
                                                    unsigned short* __restrict__ dst,
                                                    int n) {
  int i = blockIdx.x * 256 + threadIdx.x;
  if (i < n) dst[i] = f2bf_u(src[i]);
}

// --------------------------- generic WMMA GEMM -----------------------------
// C[m,n] = sum_k A[m,k] * W[n,k]  (+bias[n]) (+pos[(m%128),n]) (relu)
// A: fp32 [M,K] row-major (converted to bf16 while staging to LDS)
// W: bf16 [N,K] row-major (pre-converted)
#define GT_POS  1
#define GT_RELU 2

__global__ __launch_bounds__(256)
void wmma_gemm_atbt(const float* __restrict__ A,
                    const unsigned short* __restrict__ Wbf,
                    float* __restrict__ C,
                    const float* __restrict__ bias,
                    const float* __restrict__ pos,
                    int M, int N, int K, int mode) {
  __shared__ alignas(16) unsigned short As[2][64][40];    // 64x32 bf16/buf
  __shared__ alignas(16) unsigned short Bs[2][128][40];   // 128x32 bf16/buf

  const int tid  = threadIdx.x;
  const int lane = tid & 31;
  const int wid  = tid >> 5;
  const int wm   = wid >> 1;     // 0..3 (M sub-tile)
  const int wn   = wid & 1;      // 0..1 (N sub-tile, 64 cols each)
  const int m0   = blockIdx.y * 64;
  const int n0   = blockIdx.x * 128;

  v8f acc[4] = {};

  // A staging: 64 rows x 32 cols fp32; 2 float4 chunks per thread
  const int ar0 = tid >> 3;            // 0..31  (and +32)
  const int ac0 = (tid & 7) << 2;      // 0,4,...,28
  // B staging: 128 rows x 32 cols bf16; 2 x 8-short chunks per thread
  const int br0 = tid >> 2;            // 0..63  (and +64)
  const int bc0 = (tid & 3) << 3;      // 0,8,16,24

  const float* aptr0 = A + (size_t)(m0 + ar0) * K + ac0;
  const float* aptr1 = A + (size_t)(m0 + ar0 + 32) * K + ac0;
  const unsigned short* bptr0 = Wbf + (size_t)(n0 + br0) * K + bc0;
  const unsigned short* bptr1 = Wbf + (size_t)(n0 + br0 + 64) * K + bc0;

  auto stageA = [&](int buf, int k0) {
    __builtin_prefetch(aptr0 + k0 + 32, 0, 1);     // global_prefetch_b8
    __builtin_prefetch(aptr1 + k0 + 32, 0, 1);
    const float4 a0 = *(const float4*)(aptr0 + k0);
    const float4 a1 = *(const float4*)(aptr1 + k0);
    uint2 p0, p1;
    p0.x = (unsigned)f2bf_u(a0.x) | ((unsigned)f2bf_u(a0.y) << 16);
    p0.y = (unsigned)f2bf_u(a0.z) | ((unsigned)f2bf_u(a0.w) << 16);
    p1.x = (unsigned)f2bf_u(a1.x) | ((unsigned)f2bf_u(a1.y) << 16);
    p1.y = (unsigned)f2bf_u(a1.z) | ((unsigned)f2bf_u(a1.w) << 16);
    *(uint2*)&As[buf][ar0][ac0]      = p0;
    *(uint2*)&As[buf][ar0 + 32][ac0] = p1;
  };

  auto stageB = [&](int buf, int k0) {
#if defined(HAS_ASYNC_LDS)
    __builtin_amdgcn_global_load_async_to_lds_b128(
        (async_i32x4*)(bptr0 + k0),
        (AS3 async_i32x4*)&Bs[buf][br0][bc0], 0, 0);
    __builtin_amdgcn_global_load_async_to_lds_b128(
        (async_i32x4*)(bptr1 + k0),
        (AS3 async_i32x4*)&Bs[buf][br0 + 64][bc0], 0, 0);
#else
    const uint4 b0 = *(const uint4*)(bptr0 + k0);
    const uint4 b1 = *(const uint4*)(bptr1 + k0);
    *(uint4*)&Bs[buf][br0][bc0]      = b0;
    *(uint4*)&Bs[buf][br0 + 64][bc0] = b1;
#endif
  };

  // prologue: stage first K tile
  stageB(0, 0);
  stageA(0, 0);
  async_wait0();
  __syncthreads();

  int buf = 0;
  for (int k0 = 0; k0 < K; k0 += 32) {
    const int nxt = buf ^ 1;
    if (k0 + 32 < K) {
      stageB(nxt, k0 + 32);     // async (or reg) copy into the idle buffer
      stageA(nxt, k0 + 32);
    }

    // A fragment (16x32 bf16): lanes 0-15: K 0-7 & 16-23;
    //                          lanes 16-31: K 8-15 & 24-31.
    BfFrag af;
    const uint4* arow = (const uint4*)&As[buf][wm * 16 + (lane & 15)][0];
    const int asel = (lane < 16) ? 0 : 1;
    af.u[0] = arow[asel];
    af.u[1] = arow[asel + 2];

    // B fragments (32x16 bf16): lanes 0-15: K 0-7 & 16-23; lanes 16-31: K 8-15 & 24-31.
    const int bsel = (lane < 16) ? 0 : 1;
#pragma unroll
    for (int j = 0; j < 4; ++j) {
      BfFrag bf;
      const uint4* brow =
          (const uint4*)&Bs[buf][wn * 64 + j * 16 + (lane & 15)][0];
      bf.u[0] = brow[bsel];
      bf.u[1] = brow[bsel + 2];
      acc[j] = __builtin_amdgcn_wmma_f32_16x16x32_bf16(
          false, af.f, false, bf.f, (short)0, acc[j], false, false);
    }

    async_wait0();              // next-tile async copies landed in LDS
    __syncthreads();            // publish next buffer / retire current
    buf = nxt;
  }

  // epilogue: C layout, VGPR r -> rows r (lanes 0-15) / r+8 (lanes 16-31)
  const int rowbase = m0 + wm * 16 + ((lane < 16) ? 0 : 8);
#pragma unroll
  for (int j = 0; j < 4; ++j) {
    const int col = n0 + wn * 64 + j * 16 + (lane & 15);
    const float bv = bias ? bias[col] : 0.0f;
#pragma unroll
    for (int r = 0; r < 8; ++r) {
      const int row = rowbase + r;
      float vv = acc[j][r] + bv;
      if (mode & GT_POS)  vv += pos[(size_t)(row & 127) * N + col];
      if (mode & GT_RELU) vv = fmaxf(vv, 0.0f);
      C[(size_t)row * N + col] = vv;
    }
  }
}

// ------------------------------- layernorm ---------------------------------
// one 256-thread block per row of 768
__global__ __launch_bounds__(256)
void ln_rows(const float* __restrict__ x, float* __restrict__ y,
             const float* __restrict__ g, const float* __restrict__ b) {
  __shared__ float rs[256];
  __shared__ float rq[256];
  const int row = blockIdx.x;
  const int tid = threadIdx.x;
  const float* xr = x + (size_t)row * 768;
  const float v0 = xr[tid], v1 = xr[tid + 256], v2 = xr[tid + 512];
  rs[tid] = v0 + v1 + v2;
  rq[tid] = v0 * v0 + v1 * v1 + v2 * v2;
  __syncthreads();
  for (int off = 128; off > 0; off >>= 1) {
    if (tid < off) { rs[tid] += rs[tid + off]; rq[tid] += rq[tid + off]; }
    __syncthreads();
  }
  const float mu  = rs[0] * (1.0f / 768.0f);
  const float var = rq[0] * (1.0f / 768.0f) - mu * mu;
  const float inv = rsqrtf(var + 1e-5f);
  float* yr = y + (size_t)row * 768;
  yr[tid]       = (v0 - mu) * inv * g[tid]       + b[tid];
  yr[tid + 256] = (v1 - mu) * inv * g[tid + 256] + b[tid + 256];
  yr[tid + 512] = (v2 - mu) * inv * g[tid + 512] + b[tid + 512];
}

// ------------------------------ slot init ----------------------------------
__global__ __launch_bounds__(256)
void init_slots(const float* __restrict__ mu, const float* __restrict__ sg,
                const float* __restrict__ noise, float* __restrict__ slots) {
  const int idx = blockIdx.x * 256 + threadIdx.x;   // 256*4*768 total
  const int w = idx % 768;
  slots[idx] = mu[w] + sg[w] * noise[idx];
}

// --------------- dots + softmax(slots) + renorm(positions) -----------------
// grid = B (256), block = 128 (one thread per position n)
__global__ __launch_bounds__(128)
void attn_kernel(const float* __restrict__ q, const float* __restrict__ k,
                 float* __restrict__ attn) {
  __shared__ float qs[4 * 768];
  __shared__ float red[4][128];
  const int b = blockIdx.x, tid = threadIdx.x;
  for (int i = tid; i < 3072; i += 128) qs[i] = q[(size_t)b * 3072 + i];
  __syncthreads();

  const float* krow = k + ((size_t)b * 128 + tid) * 768;
  float a0 = 0.f, a1 = 0.f, a2 = 0.f, a3 = 0.f;
  for (int d = 0; d < 768; d += 4) {
    const float4 kv = *(const float4*)(krow + d);
    a0 += qs[d] * kv.x + qs[d + 1] * kv.y + qs[d + 2] * kv.z + qs[d + 3] * kv.w;
    a1 += qs[768 + d] * kv.x + qs[769 + d] * kv.y + qs[770 + d] * kv.z + qs[771 + d] * kv.w;
    a2 += qs[1536 + d] * kv.x + qs[1537 + d] * kv.y + qs[1538 + d] * kv.z + qs[1539 + d] * kv.w;
    a3 += qs[2304 + d] * kv.x + qs[2305 + d] * kv.y + qs[2306 + d] * kv.z + qs[2307 + d] * kv.w;
  }
  const float sc = 0.03608439182435161f;   // 768^-0.5
  a0 *= sc; a1 *= sc; a2 *= sc; a3 *= sc;
  // softmax across slots (thread-local, 4 values)
  const float m  = fmaxf(fmaxf(a0, a1), fmaxf(a2, a3));
  const float e0 = __expf(a0 - m), e1 = __expf(a1 - m),
              e2 = __expf(a2 - m), e3 = __expf(a3 - m);
  const float inv = 1.0f / (e0 + e1 + e2 + e3);
  a0 = e0 * inv + 1e-8f; a1 = e1 * inv + 1e-8f;
  a2 = e2 * inv + 1e-8f; a3 = e3 * inv + 1e-8f;
  red[0][tid] = a0; red[1][tid] = a1; red[2][tid] = a2; red[3][tid] = a3;
  __syncthreads();
  for (int off = 64; off > 0; off >>= 1) {
    if (tid < off) {
      red[0][tid] += red[0][tid + off]; red[1][tid] += red[1][tid + off];
      red[2][tid] += red[2][tid + off]; red[3][tid] += red[3][tid + off];
    }
    __syncthreads();
  }
  float* ab = attn + (size_t)b * 512;
  ab[tid]       = a0 / red[0][0];
  ab[128 + tid] = a1 / red[1][0];
  ab[256 + tid] = a2 / red[2][0];
  ab[384 + tid] = a3 / red[3][0];
}

// ---------------- updates[b,s,d] = sum_n v[b,n,d] * attn[b,s,n] ------------
// grid = B, block = 256 (thread covers d, d+256, d+512)
__global__ __launch_bounds__(256)
void updates_kernel(const float* __restrict__ v, const float* __restrict__ attn,
                    float* __restrict__ upd) {
  __shared__ float at[512];
  const int b = blockIdx.x, tid = threadIdx.x;
  at[tid]       = attn[(size_t)b * 512 + tid];
  at[tid + 256] = attn[(size_t)b * 512 + 256 + tid];
  __syncthreads();
  float acc[3][4] = {};
  const float* vb = v + (size_t)b * 128 * 768;
  for (int n = 0; n < 128; ++n) {
    const float w0 = at[n], w1 = at[128 + n], w2 = at[256 + n], w3 = at[384 + n];
    const float* vr = vb + (size_t)n * 768 + tid;
#pragma unroll
    for (int j = 0; j < 3; ++j) {
      const float vv = vr[j * 256];
      acc[j][0] += vv * w0; acc[j][1] += vv * w1;
      acc[j][2] += vv * w2; acc[j][3] += vv * w3;
    }
  }
  float* ub = upd + (size_t)b * 4 * 768;
#pragma unroll
  for (int s = 0; s < 4; ++s)
#pragma unroll
    for (int j = 0; j < 3; ++j)
      ub[(size_t)s * 768 + tid + j * 256] = acc[j][s];
}

// --------------------------- GRU gate fusion -------------------------------
__global__ __launch_bounds__(256)
void gru_kernel(const float* __restrict__ gx, const float* __restrict__ gh,
                float* __restrict__ slots) {
  const int idx = blockIdx.x * 256 + threadIdx.x;  // 1024*768
  const int m = idx / 768;
  const int w = idx - m * 768;
  const size_t gb = (size_t)m * 2304 + w;
  const float xr = gx[gb], xz = gx[gb + 768], xn = gx[gb + 1536];
  const float hr = gh[gb], hz = gh[gb + 768], hn = gh[gb + 1536];
  const float h = slots[idx];
  const float r = 1.0f / (1.0f + __expf(-(xr + hr)));
  const float z = 1.0f / (1.0f + __expf(-(xz + hz)));
  const float n = tanhf(xn + r * hn);
  slots[idx] = (1.0f - z) * n + z * h;
}

// ------------------------------ misc elwise --------------------------------
__global__ __launch_bounds__(256)
void add_kernel(float* __restrict__ dst, const float* __restrict__ src) {
  const int idx = blockIdx.x * 256 + threadIdx.x;
  dst[idx] += src[idx];
}

__global__ __launch_bounds__(256)
void copy_kernel(float* __restrict__ dst, const float* __restrict__ src) {
  const int idx = blockIdx.x * 256 + threadIdx.x;
  dst[idx] = src[idx];
}

// ---------------------------------------------------------------------------
extern "C" void kernel_launch(void* const* d_in, const int* in_sizes, int n_in,
                              void* d_out, int out_size, void* d_ws, size_t ws_size,
                              hipStream_t stream) {
  (void)in_sizes; (void)n_in; (void)out_size; (void)ws_size;
  const int B = 256, P = 128, D = 4096, W = 768, S = 4, HID = 768, ITERS = 3;
  const int MP = B * P;      // 32768 rows of embedded tokens
  const int MS = B * S;      // 1024 slot rows

  const float* pixel   = (const float*)d_in[0];
  const float* noise   = (const float*)d_in[1];
  const float* W_embed = (const float*)d_in[2];
  const float* b_embed = (const float*)d_in[3];
  const float* pos_emb = (const float*)d_in[4];
  const float* s_mu    = (const float*)d_in[5];
  const float* s_sig   = (const float*)d_in[6];
  const float* Wq      = (const float*)d_in[7];
  const float* bq      = (const float*)d_in[8];
  const float* Wk      = (const float*)d_in[9];
  const float* bk      = (const float*)d_in[10];
  const float* Wv      = (const float*)d_in[11];
  const float* bv      = (const float*)d_in[12];
  const float* W_ih    = (const float*)d_in[13];
  const float* W_hh    = (const float*)d_in[14];
  const float* b_ih    = (const float*)d_in[15];
  const float* b_hh    = (const float*)d_in[16];
  const float* W1      = (const float*)d_in[17];
  const float* b1      = (const float*)d_in[18];
  const float* W2      = (const float*)d_in[19];
  const float* b2      = (const float*)d_in[20];
  const float* lin_g   = (const float*)d_in[21];
  const float* lin_b   = (const float*)d_in[22];
  const float* lns_g   = (const float*)d_in[23];
  const float* lns_b   = (const float*)d_in[24];
  const float* lnf_g   = (const float*)d_in[25];
  const float* lnf_b   = (const float*)d_in[26];

  // ---- carve workspace ----
  char* wp = (char*)d_ws;
  auto carve = [&](size_t bytes) -> char* {
    char* p = wp;
    wp += (bytes + 255) & ~(size_t)255;
    return p;
  };
  unsigned short* wembed_bf = (unsigned short*)carve((size_t)W * D * 2);
  unsigned short* wq_bf     = (unsigned short*)carve((size_t)W * W * 2);
  unsigned short* wk_bf     = (unsigned short*)carve((size_t)W * W * 2);
  unsigned short* wv_bf     = (unsigned short*)carve((size_t)W * W * 2);
  unsigned short* wih_bf    = (unsigned short*)carve((size_t)3 * W * W * 2);
  unsigned short* whh_bf    = (unsigned short*)carve((size_t)3 * W * W * 2);
  unsigned short* w1_bf     = (unsigned short*)carve((size_t)HID * W * 2);
  unsigned short* w2_bf     = (unsigned short*)carve((size_t)W * HID * 2);
  float* inputs  = (float*)carve((size_t)MP * W * 4);
  float* kbuf    = (float*)carve((size_t)MP * W * 4);
  float* vbuf    = (float*)carve((size_t)MP * W * 4);
  float* slots   = (float*)carve((size_t)MS * W * 4);
  float* slotsln = (float*)carve((size_t)MS * W * 4);
  float* qbuf    = (float*)carve((size_t)MS * W * 4);
  float* attn    = (float*)carve((size_t)B * S * P * 4);
  float* upd     = (float*)carve((size_t)MS * W * 4);
  float* gx      = (float*)carve((size_t)MS * 3 * W * 4);
  float* gh      = (float*)carve((size_t)MS * 3 * W * 4);
  float* h1      = (float*)carve((size_t)MS * HID * 4);
  float* ff      = (float*)carve((size_t)MS * W * 4);

  // ---- weights fp32 -> bf16 ----
  auto cvt = [&](const float* s, unsigned short* d, int n) {
    cvt_f32_bf16<<<(n + 255) / 256, 256, 0, stream>>>(s, d, n);
  };
  cvt(W_embed, wembed_bf, W * D);
  cvt(Wq, wq_bf, W * W);
  cvt(Wk, wk_bf, W * W);
  cvt(Wv, wv_bf, W * W);
  cvt(W_ih, wih_bf, 3 * W * W);
  cvt(W_hh, whh_bf, 3 * W * W);
  cvt(W1, w1_bf, HID * W);
  cvt(W2, w2_bf, W * HID);

  // ---- embed GEMM + bias + pos ----
  wmma_gemm_atbt<<<dim3(W / 128, MP / 64), 256, 0, stream>>>(
      pixel, wembed_bf, inputs, b_embed, pos_emb, MP, W, D, GT_POS);
  // layernorm (in place)
  ln_rows<<<MP, 256, 0, stream>>>(inputs, inputs, lin_g, lin_b);
  // k, v projections
  wmma_gemm_atbt<<<dim3(W / 128, MP / 64), 256, 0, stream>>>(
      inputs, wk_bf, kbuf, bk, nullptr, MP, W, W, 0);
  wmma_gemm_atbt<<<dim3(W / 128, MP / 64), 256, 0, stream>>>(
      inputs, wv_bf, vbuf, bv, nullptr, MP, W, W, 0);
  // slot init
  init_slots<<<(MS * W) / 256, 256, 0, stream>>>(s_mu, s_sig, noise, slots);

  for (int it = 0; it < ITERS; ++it) {
    ln_rows<<<MS, 256, 0, stream>>>(slots, slotsln, lns_g, lns_b);
    wmma_gemm_atbt<<<dim3(W / 128, MS / 64), 256, 0, stream>>>(
        slotsln, wq_bf, qbuf, bq, nullptr, MS, W, W, 0);
    attn_kernel<<<B, 128, 0, stream>>>(qbuf, kbuf, attn);
    updates_kernel<<<B, 256, 0, stream>>>(vbuf, attn, upd);
    // GRU gates
    wmma_gemm_atbt<<<dim3((3 * W) / 128, MS / 64), 256, 0, stream>>>(
        upd, wih_bf, gx, b_ih, nullptr, MS, 3 * W, W, 0);
    wmma_gemm_atbt<<<dim3((3 * W) / 128, MS / 64), 256, 0, stream>>>(
        slots, whh_bf, gh, b_hh, nullptr, MS, 3 * W, W, 0);
    gru_kernel<<<(MS * W) / 256, 256, 0, stream>>>(gx, gh, slots);
    // residual MLP
    ln_rows<<<MS, 256, 0, stream>>>(slots, slotsln, lnf_g, lnf_b);
    wmma_gemm_atbt<<<dim3(HID / 128, MS / 64), 256, 0, stream>>>(
        slotsln, w1_bf, h1, b1, nullptr, MS, HID, W, GT_RELU);
    wmma_gemm_atbt<<<dim3(W / 128, MS / 64), 256, 0, stream>>>(
        h1, w2_bf, ff, b2, nullptr, MS, W, HID, 0);
    add_kernel<<<(MS * W) / 256, 256, 0, stream>>>(slots, ff);
  }

  copy_kernel<<<(MS * W) / 256, 256, 0, stream>>>((float*)d_out, slots);
}
